// _SelfAttention_9036611191440
// MI455X (gfx1250) — compile-verified, matrix operand order checked by reading, not run on hardware
//
#include <hip/hip_runtime.h>
#include <hip/hip_bf16.h>
#include <stdint.h>

typedef __attribute__((ext_vector_type(16))) __bf16 v16bf;
typedef __attribute__((ext_vector_type(8)))  float  v8f;

#define NUM_HEADS 16
#define DIMV      2048
#define HEAD_DIM  128
#define LSEQ      2048
#define BATCH     2

__device__ __forceinline__ unsigned short f2bf(float f) {
  __hip_bfloat16 h = __float2bfloat16(f);
  unsigned short u; __builtin_memcpy(&u, &h, 2); return u;
}
__device__ __forceinline__ float bf2f(unsigned short u) {
  __hip_bfloat16 h; __builtin_memcpy(&h, &u, 2);
  return __bfloat162float(h);
}
__device__ __forceinline__ __bf16 us2bf(unsigned short u) {
  __bf16 b; __builtin_memcpy(&b, &u, 2); return b;
}

// ---- CDNA5 async global->LDS copy (ASYNCcnt-tracked, no VGPR round trip) ----
__device__ __forceinline__ void async_b128(void* lds, const unsigned short* g) {
  unsigned off = (unsigned)(uintptr_t)lds;         // low 32 bits = LDS offset
  uint64_t ga = (uint64_t)(uintptr_t)g;
  asm volatile("global_load_async_to_lds_b128 %0, %1, off"
               :: "v"(off), "v"(ga) : "memory");
}
__device__ __forceinline__ void async_wait0() {
  asm volatile("s_wait_asynccnt 0x0" ::: "memory");
}

// A/B fragment K-index pattern for 16-bit WMMA operands (lane-half grp).
__device__ __forceinline__ int frag_k(int grp, int e) {
  return (e < 8) ? (grp * 8 + e) : (grp * 8 + e + 8);
}

// ---------------------------------------------------------------------------
// Elementwise f32 -> bf16 convert (8 elems/thread, vectorized)
// ---------------------------------------------------------------------------
__global__ __launch_bounds__(256)
void convert_bf16_kernel(const float* __restrict__ in,
                         unsigned short* __restrict__ out) {
  int i = (blockIdx.x * 256 + threadIdx.x) * 8;
  float4 f0 = *(const float4*)(in + i);
  float4 f1 = *(const float4*)(in + i + 4);
  unsigned short t[8] = { f2bf(f0.x), f2bf(f0.y), f2bf(f0.z), f2bf(f0.w),
                          f2bf(f1.x), f2bf(f1.y), f2bf(f1.z), f2bf(f1.w) };
  *(uint4*)(out + i) = *(const uint4*)t;
}

// ---------------------------------------------------------------------------
// Transpose + convert: in f32 [R][C] -> out bf16 [C][R]. 64x64 LDS tile.
// ---------------------------------------------------------------------------
__global__ __launch_bounds__(256)
void transpose_f32_bf16_kernel(const float* __restrict__ in,
                               unsigned short* __restrict__ out,
                               int R, int C) {
  __shared__ unsigned short T[64][72];
  const int tid = threadIdx.x;
  const int r0 = blockIdx.y * 64, c0 = blockIdx.x * 64;
#pragma unroll
  for (int p = 0; p < 4; ++p) {
    int item = p * 256 + tid;                 // 1024 float4 chunks
    int r = item >> 4, c = (item & 15) * 4;
    float4 f = *(const float4*)(in + (size_t)(r0 + r) * C + c0 + c);
    T[r][c + 0] = f2bf(f.x); T[r][c + 1] = f2bf(f.y);
    T[r][c + 2] = f2bf(f.z); T[r][c + 3] = f2bf(f.w);
  }
  __syncthreads();
#pragma unroll
  for (int p = 0; p < 2; ++p) {
    int item = p * 256 + tid;                 // 512 uint4 chunks
    int c = item >> 3, rr = (item & 7) * 8;
    unsigned short t[8];
#pragma unroll
    for (int j = 0; j < 8; ++j) t[j] = T[rr + j][c];
    *(uint4*)(out + (size_t)(c0 + c) * R + r0 + rr) = *(const uint4*)t;
  }
}

// ---------------------------------------------------------------------------
// bf16 slab transpose: in [Z][R][C] -> out [Z][C][R]. 64x64 LDS tile.
// ---------------------------------------------------------------------------
__global__ __launch_bounds__(256)
void transpose_bf16_kernel(const unsigned short* __restrict__ in,
                           unsigned short* __restrict__ out,
                           int R, int C) {
  __shared__ unsigned short T[64][72];
  const int tid = threadIdx.x;
  const size_t slab = (size_t)blockIdx.z * R * C;
  const int r0 = blockIdx.y * 64, c0 = blockIdx.x * 64;
#pragma unroll
  for (int p = 0; p < 2; ++p) {
    int item = p * 256 + tid;                 // 512 uint4 chunks
    int r = item >> 3, cc = (item & 7) * 8;
    *(uint4*)(&T[r][cc]) =
        *(const uint4*)(in + slab + (size_t)(r0 + r) * C + c0 + cc);
  }
  __syncthreads();
#pragma unroll
  for (int p = 0; p < 2; ++p) {
    int item = p * 256 + tid;
    int c = item >> 3, rr = (item & 7) * 8;
    unsigned short t[8];
#pragma unroll
    for (int j = 0; j < 8; ++j) t[j] = T[rr + j][c];
    *(uint4*)(out + slab + (size_t)(c0 + c) * R + r0 + rr) = *(const uint4*)t;
  }
}

// ---------------------------------------------------------------------------
// WMMA GEMM (TN): C[M,N] = A[M,K] @ Bt[N,K]^T (+bias). A,Bt bf16, K-contiguous.
// Block tile 128Mx64N, K-step 32, async double-buffered LDS staging.
// Fragments preloaded before each WMMA burst (independent accumulators).
// ---------------------------------------------------------------------------
template<bool ADD_BIAS, bool OUT_BF16>
__global__ __launch_bounds__(256)
void gemm_tn_kernel(const unsigned short* __restrict__ A,
                    const unsigned short* __restrict__ Bt,
                    const float* __restrict__ bias,
                    void* __restrict__ Cp, int M, int N, int K) {
  __shared__ unsigned short As[2][128][40];
  __shared__ unsigned short Bs[2][64][40];

  const int tid  = threadIdx.x;
  const int wave = tid >> 5, lane = tid & 31;
  const int row  = lane & 15, grp = lane >> 4;
  const int m0 = blockIdx.y * 128, n0 = blockIdx.x * 64;

  auto stage = [&](int buf, int k0) {
#pragma unroll
    for (int p = 0; p < 2; ++p) {             // A: 512 b128 chunks
      int item = p * 256 + tid;
      int r = item >> 2, c = (item & 3) * 8;
      async_b128(&As[buf][r][c], A + (size_t)(m0 + r) * K + k0 + c);
    }
    {                                          // B: 256 b128 chunks
      int r = tid >> 2, c = (tid & 3) * 8;
      async_b128(&Bs[buf][r][c], Bt + (size_t)(n0 + r) * K + k0 + c);
    }
  };

  v8f acc[4];
#pragma unroll
  for (int i = 0; i < 4; ++i) acc[i] = (v8f){};

  stage(0, 0);
  async_wait0();
  __syncthreads();

  int cur = 0;
  for (int k0 = 0; k0 < K; k0 += 32) {
    int nxt = cur ^ 1;
    if (k0 + 32 < K) stage(nxt, k0 + 32);     // overlap with WMMA below

    // preload 1 A fragment + 4 B fragments, then 4 independent WMMAs
    v16bf a, bfr[4];
#pragma unroll
    for (int e = 0; e < 16; ++e)
      a[e] = us2bf(As[cur][wave * 16 + row][frag_k(grp, e)]);
#pragma unroll
    for (int nt = 0; nt < 4; ++nt)
#pragma unroll
      for (int e = 0; e < 16; ++e)
        bfr[nt][e] = us2bf(Bs[cur][nt * 16 + row][frag_k(grp, e)]);
#pragma unroll
    for (int nt = 0; nt < 4; ++nt)
      acc[nt] = __builtin_amdgcn_wmma_f32_16x16x32_bf16(
          false, a, false, bfr[nt], (short)0, acc[nt], false, false);

    async_wait0();
    __syncthreads();
    cur = nxt;
  }

#pragma unroll
  for (int nt = 0; nt < 4; ++nt)
#pragma unroll
    for (int r8 = 0; r8 < 8; ++r8) {
      int gm = m0 + wave * 16 + grp * 8 + r8;
      int gn = n0 + nt * 16 + row;
      float v = acc[nt][r8];
      if (ADD_BIAS) v += bias[gn];
      if (OUT_BF16) ((unsigned short*)Cp)[(size_t)gm * N + gn] = f2bf(v);
      else          ((float*)Cp)[(size_t)gm * N + gn] = v;
    }
}

// ---------------------------------------------------------------------------
// RoPE + split qkv[b,l,3,h,d] bf16 -> head-major q/k/v [b,h,l,d] bf16.
// ---------------------------------------------------------------------------
__global__ __launch_bounds__(256)
void rope_split_kernel(const unsigned short* __restrict__ qkv,
                       const float* __restrict__ cosb,
                       const float* __restrict__ sinb,
                       unsigned short* __restrict__ qh,
                       unsigned short* __restrict__ kh,
                       unsigned short* __restrict__ vh) {
  int idx = blockIdx.x * blockDim.x + threadIdx.x;  // B*H*L*64 threads
  int j = idx & 63;
  int l = (idx >> 6) & (LSEQ - 1);
  int h = (idx >> 17) & (NUM_HEADS - 1);
  int b = idx >> 21;

  const size_t rowq = (size_t)(b * LSEQ + l) * (3 * DIMV);
  const int cbase = h * HEAD_DIM + j;
  const float c = cosb[l * 64 + j];
  const float s = sinb[l * 64 + j];
  const size_t o = ((size_t)(b * NUM_HEADS + h) * LSEQ + l) * HEAD_DIM + j;

  float q1 = bf2f(qkv[rowq + cbase]);
  float q2 = bf2f(qkv[rowq + cbase + 64]);
  qh[o]      = f2bf(q1 * c - q2 * s);
  qh[o + 64] = f2bf(q2 * c + q1 * s);

  float k1 = bf2f(qkv[rowq + DIMV + cbase]);
  float k2 = bf2f(qkv[rowq + DIMV + cbase + 64]);
  kh[o]      = f2bf(k1 * c - k2 * s);
  kh[o + 64] = f2bf(k2 * c + k1 * s);

  vh[o]      = qkv[rowq + 2 * DIMV + cbase];
  vh[o + 64] = qkv[rowq + 2 * DIMV + cbase + 64];
}

// ---------------------------------------------------------------------------
// Flash attention. Block = (b*H+h, 128-query tile), 8 waves x 16 queries.
// 64-key steps (amortizes online-softmax rescale over 2x keys).
// K staged [64 keys][128 d]; V staged transposed [128 d][64 keys].
// Async double-buffered K/V staging; fragments preloaded per WMMA burst.
// ---------------------------------------------------------------------------
__global__ __launch_bounds__(256)
void flash_attn_kernel(const unsigned short* __restrict__ qh,
                       const unsigned short* __restrict__ kh,
                       const unsigned short* __restrict__ vt,
                       unsigned short* __restrict__ ctx) {
  __shared__ unsigned short Qs[128][HEAD_DIM + 8];   // 34.8 KB
  __shared__ unsigned short Ks[2][64][HEAD_DIM + 8]; // 34.8 KB
  __shared__ unsigned short Vts[2][HEAD_DIM][72];    // 36.9 KB
  __shared__ unsigned short Ps[8][16 * 64];          // 16.4 KB

  const int tid  = threadIdx.x;
  const int wave = tid >> 5, lane = tid & 31;
  const int row  = lane & 15, grp = lane >> 4;
  const int bh = blockIdx.x;
  const int q0 = blockIdx.y * 128;
  const size_t head_base = (size_t)bh * LSEQ * HEAD_DIM;
  const float sscale = 0.08838834764831845f;         // 1/sqrt(128)

  // ---- async-stage Q tile 128x128 ----
#pragma unroll
  for (int p = 0; p < 8; ++p) {
    int item = p * 256 + tid;                        // 2048 b128 chunks
    int r = item >> 4, c = (item & 15) * 8;
    async_b128(&Qs[r][c], qh + head_base + (size_t)(q0 + r) * HEAD_DIM + c);
  }

  auto stageKV = [&](int buf, int kt) {
#pragma unroll
    for (int p = 0; p < 4; ++p) {
      int item = p * 256 + tid;                      // 1024 chunks each
      { int r = item >> 4, c = (item & 15) * 8;      // K: 64x128
        async_b128(&Ks[buf][r][c],
                   kh + head_base + (size_t)(kt + r) * HEAD_DIM + c); }
      { int r = item >> 3, c = (item & 7) * 8;       // Vt: 128x64
        async_b128(&Vts[buf][r][c],
                   vt + head_base + (size_t)r * LSEQ + kt + c); }
    }
  };

  stageKV(0, 0);
  async_wait0();
  __syncthreads();

  v16bf qf[4];
#pragma unroll
  for (int kc = 0; kc < 4; ++kc)
#pragma unroll
    for (int e = 0; e < 16; ++e)
      qf[kc][e] = us2bf(Qs[wave * 16 + row][kc * 32 + frag_k(grp, e)]);

  v8f o[8];
#pragma unroll
  for (int i = 0; i < 8; ++i) o[i] = (v8f){};
  float mrow[8], lrow[8];
#pragma unroll
  for (int r8 = 0; r8 < 8; ++r8) { mrow[r8] = -1e30f; lrow[r8] = 0.f; }

  int cur = 0;
  for (int kt = 0; kt < LSEQ; kt += 64) {
    int nxt = cur ^ 1;
    if (kt + 64 < LSEQ) stageKV(nxt, kt + 64);

    // ---- S = Q K^T : four 16x16 tiles over 64 keys ----
    // kc-outer / nt-inner: 4 preloaded B fragments feed 4 independent WMMAs
    v8f s4[4];
#pragma unroll
    for (int nt = 0; nt < 4; ++nt) s4[nt] = (v8f){};
#pragma unroll
    for (int kc = 0; kc < 4; ++kc) {
      v16bf bk[4];
#pragma unroll
      for (int nt = 0; nt < 4; ++nt)
#pragma unroll
        for (int e = 0; e < 16; ++e)
          bk[nt][e] = us2bf(Ks[cur][nt * 16 + row][kc * 32 + frag_k(grp, e)]);
#pragma unroll
      for (int nt = 0; nt < 4; ++nt)
        s4[nt] = __builtin_amdgcn_wmma_f32_16x16x32_bf16(
            false, qf[kc], false, bk[nt], (short)0, s4[nt], false, false);
    }

    // ---- online softmax (rows in VGPR index; columns across 16 lanes) ----
    float tmax[8];
#pragma unroll
    for (int r8 = 0; r8 < 8; ++r8) {
      float m01 = fmaxf(s4[0][r8], s4[1][r8]);
      float m23 = fmaxf(s4[2][r8], s4[3][r8]);
      tmax[r8] = fmaxf(m01, m23) * sscale;
    }
#pragma unroll
    for (int off = 1; off < 16; off <<= 1)
#pragma unroll
      for (int r8 = 0; r8 < 8; ++r8)
        tmax[r8] = fmaxf(tmax[r8], __shfl_xor(tmax[r8], off, 32));

    float corr[8], rsum[8], pr[4][8];
#pragma unroll
    for (int r8 = 0; r8 < 8; ++r8) {
      float mnew = fmaxf(mrow[r8], tmax[r8]);
      corr[r8] = __expf(mrow[r8] - mnew);
      mrow[r8] = mnew;
      float acc = 0.f;
#pragma unroll
      for (int nt = 0; nt < 4; ++nt) {
        float p = __expf(s4[nt][r8] * sscale - mnew);
        pr[nt][r8] = p;
        acc += p;
      }
      rsum[r8] = acc;
    }
#pragma unroll
    for (int off = 1; off < 16; off <<= 1)
#pragma unroll
      for (int r8 = 0; r8 < 8; ++r8)
        rsum[r8] += __shfl_xor(rsum[r8], off, 32);
#pragma unroll
    for (int r8 = 0; r8 < 8; ++r8) {
      lrow[r8] = lrow[r8] * corr[r8] + rsum[r8];
#pragma unroll
      for (int d8 = 0; d8 < 8; ++d8) o[d8][r8] *= corr[r8];
    }

    // ---- re-fragment P through per-wave LDS (C-layout -> A-layout) ----
#pragma unroll
    for (int nt = 0; nt < 4; ++nt)
#pragma unroll
      for (int r8 = 0; r8 < 8; ++r8)
        Ps[wave][(grp * 8 + r8) * 64 + nt * 16 + row] = f2bf(pr[nt][r8]);
    __syncthreads();

    v16bf pa0, pa1;
#pragma unroll
    for (int e = 0; e < 16; ++e) {
      int kk = frag_k(grp, e);
      pa0[e] = us2bf(Ps[wave][row * 64 + kk]);
      pa1[e] = us2bf(Ps[wave][row * 64 + 32 + kk]);
    }
    // ---- O += P @ V (V transposed in LDS: contiguous along keys) ----
#pragma unroll
    for (int d8 = 0; d8 < 8; ++d8) {
      v16bf bv0, bv1;
#pragma unroll
      for (int e = 0; e < 16; ++e) {
        int kk = frag_k(grp, e);
        bv0[e] = us2bf(Vts[cur][d8 * 16 + row][kk]);
        bv1[e] = us2bf(Vts[cur][d8 * 16 + row][32 + kk]);
      }
      o[d8] = __builtin_amdgcn_wmma_f32_16x16x32_bf16(
          false, pa0, false, bv0, (short)0, o[d8], false, false);
      o[d8] = __builtin_amdgcn_wmma_f32_16x16x32_bf16(
          false, pa1, false, bv1, (short)0, o[d8], false, false);
    }
    async_wait0();
    __syncthreads();
    cur = nxt;
  }

  const int b = bh >> 4, h = bh & 15;
#pragma unroll
  for (int d8 = 0; d8 < 8; ++d8)
#pragma unroll
    for (int r8 = 0; r8 < 8; ++r8) {
      int gm = q0 + wave * 16 + grp * 8 + r8;
      int gn = h * HEAD_DIM + d8 * 16 + row;
      ctx[(size_t)(b * LSEQ + gm) * DIMV + gn] = f2bf(o[d8][r8] / lrow[r8]);
    }
}

// ---------------------------------------------------------------------------
extern "C" void kernel_launch(void* const* d_in, const int* in_sizes, int n_in,
                              void* d_out, int out_size, void* d_ws, size_t ws_size,
                              hipStream_t stream) {
  (void)in_sizes; (void)n_in; (void)out_size; (void)ws_size;
  const float* x     = (const float*)d_in[0];
  const float* cosb  = (const float*)d_in[1];
  const float* sinb  = (const float*)d_in[2];
  const float* w_qkv = (const float*)d_in[3];
  const float* w_out = (const float*)d_in[4];
  const float* b_out = (const float*)d_in[5];
  float* out = (float*)d_out;

  const size_t x_elems    = (size_t)BATCH * LSEQ * DIMV;
  const size_t wqkv_elems = (size_t)DIMV * 3 * DIMV;
  const size_t wout_elems = (size_t)DIMV * DIMV;
  const size_t qkv_elems  = (size_t)BATCH * LSEQ * 3 * DIMV;
  const size_t head_elems = (size_t)BATCH * NUM_HEADS * LSEQ * HEAD_DIM;

  unsigned short* xh    = (unsigned short*)d_ws;
  unsigned short* wqkvT = xh + x_elems;
  unsigned short* woutT = wqkvT + wqkv_elems;
  unsigned short* qkvh  = woutT + wout_elems;
  unsigned short* qh    = qkvh + qkv_elems;
  unsigned short* kh    = qh + head_elems;
  unsigned short* vh    = kh + head_elems;
  unsigned short* vt    = qkvh;   // qkv dead after rope_split
  unsigned short* ctx   = xh;     // x-bf16 dead after GEMM1

  const int Mrows = BATCH * LSEQ;  // 4096

  // 0) precision/layout prep
  convert_bf16_kernel<<<x_elems / 2048, 256, 0, stream>>>(x, xh);
  { dim3 g((3 * DIMV) / 64, DIMV / 64);
    transpose_f32_bf16_kernel<<<g, 256, 0, stream>>>(w_qkv, wqkvT, DIMV, 3 * DIMV); }
  { dim3 g(DIMV / 64, DIMV / 64);
    transpose_f32_bf16_kernel<<<g, 256, 0, stream>>>(w_out, woutT, DIMV, DIMV); }

  // 1) qkv = x @ w_qkv   (bf16 TN GEMM -> bf16)
  { dim3 g((3 * DIMV) / 64, Mrows / 128);
    gemm_tn_kernel<false, true><<<g, 256, 0, stream>>>(
        xh, wqkvT, nullptr, (void*)qkvh, Mrows, 3 * DIMV, DIMV); }

  // 2) RoPE + head-major split
  rope_split_kernel<<<(BATCH * NUM_HEADS * LSEQ * 64) / 256, 256, 0, stream>>>(
      qkvh, cosb, sinb, qh, kh, vh);

  // 3) per-head V transpose [l][d] -> [d][l]
  { dim3 g(HEAD_DIM / 64, LSEQ / 64, BATCH * NUM_HEADS);
    transpose_bf16_kernel<<<g, 256, 0, stream>>>(vh, vt, LSEQ, HEAD_DIM); }

  // 4) flash attention -> ctx bf16 [B, L, dim]
  { dim3 g(BATCH * NUM_HEADS, LSEQ / 128);
    flash_attn_kernel<<<g, 256, 0, stream>>>(qh, kh, vt, ctx); }

  // 5) out = ctx @ w_out + b_out  (bf16 TN GEMM -> f32)
  { dim3 g(DIMV / 64, Mrows / 128);
    gemm_tn_kernel<true, false><<<g, 256, 0, stream>>>(
        ctx, woutT, b_out, (void*)out, Mrows, DIMV, DIMV); }
}
